// Block_37658273251484
// MI455X (gfx1250) — matrix operand order repaired and checked by reading, not compiled, not run on hardware
//
#include <hip/hip_runtime.h>
#include <hip/hip_bf16.h>
#include <math.h>

// ---------------------------------------------------------------------------
// GPT block forward for MI455X (gfx1250, wave32, WMMA + Tensor Data Mover).
// GEMM/attention matmuls: v_wmma_f32_16x16x32_bf16.
// Global->LDS tiles: tensor_load_to_lds (TDM), double buffered, TENSORcnt.
// ---------------------------------------------------------------------------

typedef __attribute__((ext_vector_type(16))) __bf16 bf16x16;  // 8 VGPRs
typedef __attribute__((ext_vector_type(8)))  float  f32x8;    // 8 VGPRs
typedef __attribute__((ext_vector_type(4)))  unsigned u32x4;  // 16B vector
typedef __attribute__((ext_vector_type(8)))  int      i32x8;
typedef __attribute__((ext_vector_type(4)))  int      i32x4;

static constexpr int Bb = 2, Ts = 2048, Cc = 1024, Hh = 16, HD = 64;
static constexpr int Mtok = Bb * Ts;           // 4096 tokens
static constexpr int C3 = 3 * Cc;              // 3072

__device__ inline f32x8 zero8() {
  f32x8 z;
#pragma unroll
  for (int i = 0; i < 8; ++i) z[i] = 0.0f;
  return z;
}

union FragU { u32x4 u[2]; bf16x16 v; };
union Pack8 { u32x4 u; __bf16 e[8]; };

__device__ inline bf16x16 load_frag(const __bf16* p0, const __bf16* p1) {
  FragU f;
  f.u[0] = *(const u32x4*)p0;
  f.u[1] = *(const u32x4*)p1;
  return f.v;
}

__device__ inline f32x8 wmma_bf16(bf16x16 a, bf16x16 b, f32x8 c) {
  return __builtin_amdgcn_wmma_f32_16x16x32_bf16(false, a, false, b, (short)0, c,
                                                 false, false);
}

// ---------------------------------------------------------------------------
// TDM: 2D tile (bf16) global -> LDS, per CDNA5 D# layout (ISA ch.10.8).
// tile_d0 elements contiguous along dim0; rows advance by stride0 elements.
// LDS receives the tile packed row-major: tile_d1 rows of tile_d0 bf16.
// ---------------------------------------------------------------------------
__device__ inline void tdm_load_2d_bf16(unsigned lds_off, const void* gaddr,
                                        unsigned tile_d0, unsigned tile_d1,
                                        unsigned tensor_d0, unsigned tensor_d1,
                                        unsigned long long stride0) {
  unsigned long long ga = (unsigned long long)(uintptr_t)gaddr;
  u32x4 g0;
  g0[0] = 1u;                                    // count=1 (user descriptor)
  g0[1] = lds_off;                               // lds_addr [63:32]
  g0[2] = (unsigned)ga;                          // global_addr [95:64]
  g0[3] = (unsigned)(ga >> 32) | (2u << 30);     // global_addr hi | type=2
  i32x8 g1;
  g1[0] = (int)(1u << 16);                                        // data_size=1 -> 2B
  g1[1] = (int)((tensor_d0 & 0xffffu) << 16);                     // tensor_dim0 lo
  g1[2] = (int)((tensor_d0 >> 16) | ((tensor_d1 & 0xffffu) << 16));
  g1[3] = (int)((tensor_d1 >> 16) | (tile_d0 << 16));             // tile_dim0
  g1[4] = (int)tile_d1;                                           // tile_dim1 (dim2=0)
  g1[5] = (int)(stride0 & 0xffffffffull);                         // dim0 stride lo
  g1[6] = (int)((stride0 >> 32) & 0xffffull);                     // dim0 stride hi
  g1[7] = 0;
  i32x4 gz4 = {0, 0, 0, 0};
  i32x8 gz8 = {0, 0, 0, 0, 0, 0, 0, 0};
  // amdgpu-toolchain / clang-23 form: (g0, g1, g2, g3, g_extra, cpol)
  __builtin_amdgcn_tensor_load_to_lds(g0, g1, gz4, gz4, gz8, 0);
}

__device__ inline unsigned lds_offset(const void* p) {
  // Flat LDS aperture: addr[31:0] is the LDS byte offset.
  return (unsigned)(uintptr_t)p;
}

// ---------------------------------------------------------------------------
// Cast + transpose weights: src f32 [K][N] -> dst bf16 [N][K].
// Makes GEMM B-tiles directly TDM-loadable in the WMMA-fragment LDS layout.
// ---------------------------------------------------------------------------
__global__ void __launch_bounds__(256)
cast_transpose_bf16(const float* __restrict__ src, __bf16* __restrict__ dst,
                    int K, int N) {
  __shared__ float tile[32][33];
  int k0 = blockIdx.y * 32, n0 = blockIdx.x * 32;
  int tx = threadIdx.x & 31, ty = threadIdx.x >> 5;  // 32 x 8
#pragma unroll
  for (int i = 0; i < 4; ++i) {
    int k = ty + i * 8;
    tile[k][tx] = src[(size_t)(k0 + k) * N + n0 + tx];
  }
  __syncthreads();
#pragma unroll
  for (int i = 0; i < 4; ++i) {
    int n = ty + i * 8;
    dst[(size_t)(n0 + n) * K + k0 + tx] = (__bf16)tile[tx][n];
  }
}

// ---------------------------------------------------------------------------
// LayerNorm over C=1024, one block (256 threads) per row, bf16 output
// ---------------------------------------------------------------------------
__global__ void __launch_bounds__(256)
layernorm_to_bf16(const float* __restrict__ x, const float* __restrict__ gamma,
                  const float* __restrict__ beta, __bf16* __restrict__ out) {
  int row = blockIdx.x;
  const float* xr = x + (size_t)row * Cc;
  int tid = threadIdx.x;
  float v[4], s = 0.0f, s2 = 0.0f;
#pragma unroll
  for (int i = 0; i < 4; ++i) {
    v[i] = xr[tid + 256 * i];
    s += v[i];
    s2 += v[i] * v[i];
  }
#pragma unroll
  for (int off = 16; off > 0; off >>= 1) {
    s  += __shfl_xor(s, off);
    s2 += __shfl_xor(s2, off);
  }
  __shared__ float ws1[8], ws2[8];
  int wave = tid >> 5, lane = tid & 31;
  if (lane == 0) { ws1[wave] = s; ws2[wave] = s2; }
  __syncthreads();
  if (tid == 0) {
    float a = 0.0f, b = 0.0f;
#pragma unroll
    for (int i = 0; i < 8; ++i) { a += ws1[i]; b += ws2[i]; }
    ws1[0] = a; ws2[0] = b;
  }
  __syncthreads();
  float mean = ws1[0] * (1.0f / Cc);
  float var  = ws2[0] * (1.0f / Cc) - mean * mean;
  float rstd = rsqrtf(var + 1e-6f);
  __bf16* orow = out + (size_t)row * Cc;
#pragma unroll
  for (int i = 0; i < 4; ++i) {
    int c = tid + 256 * i;
    orow[c] = (__bf16)((v[i] - mean) * rstd * gamma[c] + beta[c]);
  }
}

// ---------------------------------------------------------------------------
// bf16 WMMA GEMM with TDM-fed double-buffered LDS tiles.
//   out = epilogue(A[M,K] @ Bt[N,K]^T + bias)
// Block tile 128x64, BK=32, 256 threads = 8 waves (4x2), 32x32 per wave.
// EPI: 0 = bias -> bf16, 1 = bias+gelu -> bf16, 2 = bias+resid -> f32
// ---------------------------------------------------------------------------
template <int EPI>
__global__ void __launch_bounds__(256)
gemm_bf16_wmma(const __bf16* __restrict__ A, const __bf16* __restrict__ Bt,
               const float* __restrict__ bias, const float* __restrict__ resid,
               __bf16* __restrict__ outb, float* __restrict__ outf,
               int M, int N, int K) {
  __shared__ __bf16 As[2][128 * 32];   // [m][k], row stride 32
  __shared__ __bf16 Bs[2][64 * 32];    // [n][k], row stride 32

  int tid = threadIdx.x;
  int wave = tid >> 5, lane = tid & 31;
  int wm = wave >> 1, wn = wave & 1;
  int m0 = blockIdx.y * 128, n0 = blockIdx.x * 64;
  int lhi = lane >> 4, llo = lane & 15;
  bool issuer = (wave == 0);

  unsigned aoff[2] = { lds_offset(&As[0][0]), lds_offset(&As[1][0]) };
  unsigned boff[2] = { lds_offset(&Bs[0][0]), lds_offset(&Bs[1][0]) };
  const __bf16* Ab = A + (size_t)m0 * K;
  const __bf16* Bb_ = Bt + (size_t)n0 * K;

  f32x8 acc[2][2];
#pragma unroll
  for (int i = 0; i < 2; ++i)
#pragma unroll
    for (int j = 0; j < 2; ++j) acc[i][j] = zero8();

  int nk = K / 32;
  if (issuer) {
    tdm_load_2d_bf16(aoff[0], Ab,  32, 128, (unsigned)K, (unsigned)M, (unsigned long long)K);
    tdm_load_2d_bf16(boff[0], Bb_, 32,  64, (unsigned)K, (unsigned)N, (unsigned long long)K);
  }
  for (int i = 0; i < nk; ++i) {
    int cur = i & 1;
    if (issuer) {
      if (i + 1 < nk) {
        int kk = (i + 1) * 32;
        tdm_load_2d_bf16(aoff[cur ^ 1], Ab + kk,  32, 128, (unsigned)K, (unsigned)M,
                         (unsigned long long)K);
        tdm_load_2d_bf16(boff[cur ^ 1], Bb_ + kk, 32,  64, (unsigned)K, (unsigned)N,
                         (unsigned long long)K);
        __builtin_amdgcn_s_wait_tensorcnt(2);   // tile i landed; i+1 in flight
      } else {
        __builtin_amdgcn_s_wait_tensorcnt(0);
      }
    }
    __syncthreads();   // publish tile i to all waves

    bf16x16 Af[2], Bf[2];
#pragma unroll
    for (int im = 0; im < 2; ++im) {
      const __bf16* base = &As[cur][(wm * 32 + im * 16 + llo) * 32];
      Af[im] = load_frag(base + 8 * lhi, base + 16 + 8 * lhi);
    }
#pragma unroll
    for (int in = 0; in < 2; ++in) {
      const __bf16* base = &Bs[cur][(wn * 32 + in * 16 + llo) * 32];
      Bf[in] = load_frag(base + 8 * lhi, base + 16 + 8 * lhi);
    }
#pragma unroll
    for (int im = 0; im < 2; ++im)
#pragma unroll
      for (int in = 0; in < 2; ++in)
        acc[im][in] = wmma_bf16(Af[im], Bf[in], acc[im][in]);
    __syncthreads();   // all done reading buf[cur] before it is re-filled
  }

#pragma unroll
  for (int im = 0; im < 2; ++im)
#pragma unroll
    for (int in = 0; in < 2; ++in)
#pragma unroll
      for (int r = 0; r < 8; ++r) {
        int row = m0 + wm * 32 + im * 16 + r + 8 * lhi;
        int col = n0 + wn * 32 + in * 16 + llo;
        float v = acc[im][in][r] + bias[col];
        if (EPI == 1) {
          float u = 0.7978845608028654f * (v + 0.044715f * v * v * v);
          v = 0.5f * v * (1.0f + tanhf(u));
        }
        if (EPI == 2) {
          v += resid[(size_t)row * N + col];
          outf[(size_t)row * N + col] = v;
        } else {
          outb[(size_t)row * N + col] = (__bf16)v;
        }
      }
}

// ---------------------------------------------------------------------------
// Flash-style causal attention. Grid: (T/64, B*H). 128 threads = 4 waves,
// each wave owns 16 query rows; 32 keys per iteration; K tile via TDM.
// qkv layout per token: [q(1024) | k(1024) | v(1024)], head h at h*64.
// ---------------------------------------------------------------------------
__global__ void __launch_bounds__(128)
attn_causal_wmma(const __bf16* __restrict__ qkv, __bf16* __restrict__ o) {
  __shared__ __bf16 Kt[32 * 64];        // [key][hd]  (TDM-filled)
  __shared__ __bf16 Vt[64 * 32];        // transposed: [hd][key]
  __shared__ __bf16 Pb[4][16 * 32];     // per-wave P scratch [row][key]

  int tid = threadIdx.x;
  int wave = tid >> 5, lane = tid & 31;
  int lhi = lane >> 4, llo = lane & 15;
  int qb = blockIdx.x * 64;
  int bh = blockIdx.y;
  int b = bh >> 4, h = bh & 15;
  size_t tok0 = (size_t)b * Ts;
  int qr0 = qb + wave * 16;
  unsigned ktoff = lds_offset(&Kt[0]);
  bool issuer = (wave == 0);

  const __bf16* qp = qkv + (tok0 + qr0 + llo) * C3 + h * HD;
  bf16x16 Qf[2];
  Qf[0] = load_frag(qp + 8 * lhi,      qp + 16 + 8 * lhi);
  Qf[1] = load_frag(qp + 32 + 8 * lhi, qp + 48 + 8 * lhi);

  f32x8 Oc[4];
#pragma unroll
  for (int j = 0; j < 4; ++j) Oc[j] = zero8();
  float m_i[8], l_i[8];
#pragma unroll
  for (int r = 0; r < 8; ++r) { m_i[r] = -1e30f; l_i[r] = 0.0f; }

  int nkb = (qb + 64) / 32;  // uniform across block
  for (int kb = 0; kb < nkb; ++kb) {
    int ks = kb * 32;
    // K tile 32x64 via TDM (rows stride C3 in the qkv tensor).
    if (issuer) {
      tdm_load_2d_bf16(ktoff, qkv + (tok0 + ks) * C3 + Cc + h * HD,
                       64, 32, (unsigned)C3, (unsigned)Ts, (unsigned long long)C3);
    }
    // V tile: vector loads + transpose into Vt[hd][key].
#pragma unroll
    for (int i = 0; i < 2; ++i) {
      int c = tid + 128 * i;
      int row = c >> 3, ch8 = (c & 7) * 8;
      const __bf16* vg = qkv + (tok0 + ks + row) * C3 + 2 * Cc + h * HD + ch8;
      Pack8 vv; vv.u = *(const u32x4*)vg;
#pragma unroll
      for (int j = 0; j < 8; ++j) Vt[(ch8 + j) * 32 + row] = vv.e[j];
    }
    if (issuer) __builtin_amdgcn_s_wait_tensorcnt(0);
    __syncthreads();

    // S = Q @ K^T (two 16x16 tiles over 32 keys; hd contraction = 2 WMMAs)
    float sv[2][8];
#pragma unroll
    for (int kt = 0; kt < 2; ++kt) {
      const __bf16* kbase = &Kt[(kt * 16 + llo) * 64];
      bf16x16 Bk0 = load_frag(kbase + 8 * lhi,      kbase + 16 + 8 * lhi);
      bf16x16 Bk1 = load_frag(kbase + 32 + 8 * lhi, kbase + 48 + 8 * lhi);
      f32x8 c = zero8();
      c = wmma_bf16(Qf[0], Bk0, c);
      c = wmma_bf16(Qf[1], Bk1, c);
#pragma unroll
      for (int r = 0; r < 8; ++r) {
        int kg = ks + kt * 16 + llo;
        int qi = qr0 + r + 8 * lhi;
        float s = c[r] * 0.125f;               // 1/sqrt(64)
        sv[kt][r] = (kg <= qi) ? s : -1e30f;   // causal mask
      }
    }

    // Online softmax: row stats shared by 16-lane half-wave groups.
    float fscale[8];
#pragma unroll
    for (int r = 0; r < 8; ++r) {
      float rm = fmaxf(sv[0][r], sv[1][r]);
#pragma unroll
      for (int off = 1; off < 16; off <<= 1) rm = fmaxf(rm, __shfl_xor(rm, off));
      float mn = fmaxf(m_i[r], rm);
      float f  = __expf(m_i[r] - mn);
      float p0 = __expf(sv[0][r] - mn);
      float p1 = __expf(sv[1][r] - mn);
      float rs = p0 + p1;
#pragma unroll
      for (int off = 1; off < 16; off <<= 1) rs += __shfl_xor(rs, off);
      l_i[r] = l_i[r] * f + rs;
      m_i[r] = mn;
      fscale[r] = f;
      sv[0][r] = p0; sv[1][r] = p1;
    }
#pragma unroll
    for (int j = 0; j < 4; ++j)
#pragma unroll
      for (int r = 0; r < 8; ++r) Oc[j][r] *= fscale[r];

    // Transpose P (C-layout -> A-layout) through per-wave LDS scratch.
    __bf16* pb = Pb[wave];
#pragma unroll
    for (int kt = 0; kt < 2; ++kt)
#pragma unroll
      for (int r = 0; r < 8; ++r)
        pb[(r + 8 * lhi) * 32 + kt * 16 + llo] = (__bf16)sv[kt][r];
    asm volatile("s_wait_dscnt 0" ::: "memory");  // same-wave DS ops in order

    const __bf16* prow = pb + llo * 32;
    bf16x16 Pf = load_frag(prow + 8 * lhi, prow + 16 + 8 * lhi);

    // O += P @ V  (4 hd tiles of 16)
#pragma unroll
    for (int j = 0; j < 4; ++j) {
      const __bf16* vb = &Vt[(j * 16 + llo) * 32];
      bf16x16 Vf = load_frag(vb + 8 * lhi, vb + 16 + 8 * lhi);
      Oc[j] = wmma_bf16(Pf, Vf, Oc[j]);
    }
    __syncthreads();
  }

#pragma unroll
  for (int r = 0; r < 8; ++r) {
    float inv = 1.0f / l_i[r];
    int tokrow = qr0 + r + 8 * lhi;
    __bf16* orow = o + (tok0 + tokrow) * Cc + h * HD;
#pragma unroll
    for (int j = 0; j < 4; ++j) orow[j * 16 + llo] = (__bf16)(Oc[j][r] * inv);
  }
}

// ---------------------------------------------------------------------------
// Host-side pipeline
// ---------------------------------------------------------------------------
extern "C" void kernel_launch(void* const* d_in, const int* in_sizes, int n_in,
                              void* d_out, int out_size, void* d_ws, size_t ws_size,
                              hipStream_t stream) {
  (void)in_sizes; (void)n_in; (void)out_size; (void)ws_size;
  const float* x     = (const float*)d_in[0];
  // d_in[1] = mask (unused: causal structure is known)
  const float* ln1_s = (const float*)d_in[2];
  const float* ln1_b = (const float*)d_in[3];
  const float* w_qkv = (const float*)d_in[4];
  const float* b_qkv = (const float*)d_in[5];
  const float* w_ap  = (const float*)d_in[6];
  const float* b_ap  = (const float*)d_in[7];
  const float* ln2_s = (const float*)d_in[8];
  const float* ln2_b = (const float*)d_in[9];
  const float* w_fc  = (const float*)d_in[10];
  const float* b_fc  = (const float*)d_in[11];
  const float* w_mp  = (const float*)d_in[12];
  const float* b_mp  = (const float*)d_in[13];

  char* ws = (char*)d_ws;
  size_t off = 0;
  auto alloc = [&](size_t bytes) -> void* {
    void* p = ws + off;
    off = (off + bytes + 255) & ~(size_t)255;
    return p;
  };
  // Transposed bf16 weights: Wt[N][K]
  __bf16* wqkv_t = (__bf16*)alloc((size_t)Cc * C3 * 2);
  __bf16* wap_t  = (__bf16*)alloc((size_t)Cc * Cc * 2);
  __bf16* wfc_t  = (__bf16*)alloc((size_t)Cc * 4 * Cc * 2);
  __bf16* wmp_t  = (__bf16*)alloc((size_t)4 * Cc * Cc * 2);
  __bf16* h_bf   = (__bf16*)alloc((size_t)Mtok * Cc * 2);
  __bf16* big_bf = (__bf16*)alloc((size_t)Mtok * 4 * Cc * 2);  // qkv, then fc
  __bf16* o_bf   = (__bf16*)alloc((size_t)Mtok * Cc * 2);
  float*  x1     = (float*) alloc((size_t)Mtok * Cc * 4);
  __bf16* qkv_bf = big_bf;
  __bf16* fc_bf  = big_bf;

  // 1) Cast + transpose weights to bf16 [N][K].
  cast_transpose_bf16<<<dim3(C3 / 32, Cc / 32), 256, 0, stream>>>(w_qkv, wqkv_t, Cc, C3);
  cast_transpose_bf16<<<dim3(Cc / 32, Cc / 32), 256, 0, stream>>>(w_ap,  wap_t,  Cc, Cc);
  cast_transpose_bf16<<<dim3(4 * Cc / 32, Cc / 32), 256, 0, stream>>>(w_fc, wfc_t, Cc, 4 * Cc);
  cast_transpose_bf16<<<dim3(Cc / 32, 4 * Cc / 32), 256, 0, stream>>>(w_mp, wmp_t, 4 * Cc, Cc);

  // 2) LN1 -> bf16
  layernorm_to_bf16<<<Mtok, 256, 0, stream>>>(x, ln1_s, ln1_b, h_bf);

  // 3) QKV GEMM: [4096,1024] @ [1024,3072] -> bf16
  gemm_bf16_wmma<0><<<dim3(C3 / 64, Mtok / 128), 256, 0, stream>>>(
      h_bf, wqkv_t, b_qkv, nullptr, qkv_bf, nullptr, Mtok, C3, Cc);

  // 4) Causal flash attention -> bf16 o
  attn_causal_wmma<<<dim3(Ts / 64, Bb * Hh), 128, 0, stream>>>(qkv_bf, o_bf);

  // 5) Attention proj + residual: x1 = x + o @ Wap + b
  gemm_bf16_wmma<2><<<dim3(Cc / 64, Mtok / 128), 256, 0, stream>>>(
      o_bf, wap_t, b_ap, x, nullptr, x1, Mtok, Cc, Cc);

  // 6) LN2 -> bf16
  layernorm_to_bf16<<<Mtok, 256, 0, stream>>>(x1, ln2_s, ln2_b, h_bf);

  // 7) FC GEMM + GELU: [4096,1024] @ [1024,4096] -> bf16
  gemm_bf16_wmma<1><<<dim3(4 * Cc / 64, Mtok / 128), 256, 0, stream>>>(
      h_bf, wfc_t, b_fc, nullptr, fc_bf, nullptr, Mtok, 4 * Cc, Cc);

  // 8) MLP proj + residual -> f32 d_out
  gemm_bf16_wmma<2><<<dim3(Cc / 64, Mtok / 128), 256, 0, stream>>>(
      fc_bf, wmp_t, b_mp, x1, nullptr, (float*)d_out, Mtok, Cc, 4 * Cc);
}
